// ShawAttention_74861279969438
// MI455X (gfx1250) — matrix operand order repaired and checked by reading, not compiled, not run on hardware
//
#include <hip/hip_runtime.h>
#include <hip/hip_bf16.h>
#include <math.h>

typedef __attribute__((ext_vector_type(16))) __bf16 v16bf;
typedef __attribute__((ext_vector_type(8)))  float  v8f;
typedef int v4i __attribute__((vector_size(16)));

#define WMMA_BF16(a, b, c) \
  __builtin_amdgcn_wmma_f32_16x16x32_bf16(false, (a), false, (b), (short)0, (c), false, false)

#if defined(__AMDGCN__) && __has_builtin(__builtin_amdgcn_global_load_async_to_lds_b128) && \
    __has_builtin(__builtin_amdgcn_s_wait_asynccnt)
#define HAVE_ASYNC_LDS 1
#else
#define HAVE_ASYNC_LDS 0
#endif

#define AS1 __attribute__((address_space(1)))
#define AS3 __attribute__((address_space(3)))

// copy 16 bytes global -> LDS (async path: GLOBAL_LOAD_ASYNC_TO_LDS_B128, ASYNCcnt)
__device__ __forceinline__ void cp16(void* lds, const void* g) {
#if HAVE_ASYNC_LDS
  __builtin_amdgcn_global_load_async_to_lds_b128((AS1 v4i*)(g), (AS3 v4i*)(lds), 0, 0);
#else
  *(uint4*)lds = *(const uint4*)g;
#endif
}
__device__ __forceinline__ void cp_wait_all() {
#if HAVE_ASYNC_LDS
  __builtin_amdgcn_s_wait_asynccnt(0);
#endif
}

constexpr int BATCH = 8, SEQ = 1024, DM = 1024, NH = 16, DK = 64;
constexpr int HG = 8;   // heads per attention block (one batch split in 2)
constexpr int IT = 16;  // query-row tile
constexpr int JT = 16;  // key-column tile

// ---------------- CDNA5 WMMA fragment loaders (contiguous k = 16*half + s, same order in A and B) ----------------
// A 16x32 bf16: lane m=lane&15 holds row m; VGPR v pair k = half*16+2v
__device__ __forceinline__ v16bf frag_a(const __bf16* p, int ld, int rowLimit, int kLimit) {
  const int lane = threadIdx.x & 31, m = lane & 15, half = lane >> 4;
  v16bf a;
  #pragma unroll
  for (int v = 0; v < 8; ++v) {
    const int k = half * 16 + 2 * v;
    const bool rok = (m < rowLimit);
    a[2 * v]     = (rok && (k     < kLimit)) ? p[m * ld + k]     : (__bf16)0.0f;
    a[2 * v + 1] = (rok && (k + 1 < kLimit)) ? p[m * ld + k + 1] : (__bf16)0.0f;
  }
  return a;
}
// B 32x16 bf16 stored transposed [n][k] (e.g. K tile [j][d]); k = half*16 + 2v
__device__ __forceinline__ v16bf frag_b_nk(const __bf16* p, int ld, int kLimit) {
  const int lane = threadIdx.x & 31, n = lane & 15, half = lane >> 4;
  v16bf b;
  #pragma unroll
  for (int v = 0; v < 8; ++v) {
    const int k = half * 16 + 2 * v;
    b[2 * v]     = (k     < kLimit) ? p[n * ld + k]     : (__bf16)0.0f;
    b[2 * v + 1] = (k + 1 < kLimit) ? p[n * ld + k + 1] : (__bf16)0.0f;
  }
  return b;
}
// B 32x16 bf16 stored [k][n] (e.g. V tile [j][d] with n over d)
__device__ __forceinline__ v16bf frag_b_kn(const __bf16* p, int ld, int kLimit) {
  const int lane = threadIdx.x & 31, n = lane & 15, half = lane >> 4;
  v16bf b;
  #pragma unroll
  for (int v = 0; v < 8; ++v) {
    const int k = half * 16 + 2 * v;
    b[2 * v]     = (k     < kLimit) ? p[k * ld + n]       : (__bf16)0.0f;
    b[2 * v + 1] = (k + 1 < kLimit) ? p[(k + 1) * ld + n] : (__bf16)0.0f;
  }
  return b;
}

// ---------------- GEMM: Y[m,n] = sum_k A[m,k] * W[n,k]  (bf16 WMMA, f32 accum) -------
// MODE 0: store bf16 into head-major [B][NH][SEQ][DK] (n -> h*64+d, m -> b*SEQ+l)
// MODE 1: store f32 row-major [m][n]
template <typename TA, int MODE>
__global__ __launch_bounds__(256) void gemm_bf16(const TA* __restrict__ A,
                                                 const float* __restrict__ W,
                                                 void* __restrict__ Out) {
  __shared__ __bf16 As[128][32];
  __shared__ __bf16 Bs[32][64];
  const int t = threadIdx.x;
  const int lane = t & 31, w = t >> 5;
  const int wm = w & 3, wn = w >> 2;
  const int half = lane >> 4, ln = lane & 15;
  const int m0 = blockIdx.y * 128, n0 = blockIdx.x * 64;
  v8f acc[2][2] = {};

  for (int k0 = 0; k0 < DM; k0 += 32) {
    __syncthreads();
    {
      const int m = t >> 1, kh = (t & 1) * 16;
      const TA* src = A + (size_t)(m0 + m) * DM + k0 + kh;
      #pragma unroll
      for (int q = 0; q < 16; ++q) As[m][kh + q] = (__bf16)src[q];
      const int n = t & 63, ks = (t >> 6) * 8;
      const float* wsrc = W + (size_t)(n0 + n) * DM + k0 + ks;
      #pragma unroll
      for (int q = 0; q < 8; ++q) Bs[ks + q][n] = (__bf16)wsrc[q];
      if (k0 + 32 < DM) {  // global_prefetch_b8 for the next k-slab
        __builtin_prefetch(src + 32, 0, 1);
        __builtin_prefetch(wsrc + 32, 0, 1);
      }
    }
    __syncthreads();
    v16bf af[2], bfr[2];
    #pragma unroll
    for (int mi = 0; mi < 2; ++mi) af[mi] = frag_a(&As[wm * 32 + mi * 16][0], 32, 16, 32);
    #pragma unroll
    for (int ni = 0; ni < 2; ++ni) bfr[ni] = frag_b_kn(&Bs[0][wn * 32 + ni * 16], 64, 32);
    #pragma unroll
    for (int mi = 0; mi < 2; ++mi)
      #pragma unroll
      for (int ni = 0; ni < 2; ++ni) acc[mi][ni] = WMMA_BF16(af[mi], bfr[ni], acc[mi][ni]);
  }

  #pragma unroll
  for (int mi = 0; mi < 2; ++mi)
    #pragma unroll
    for (int ni = 0; ni < 2; ++ni)
      #pragma unroll
      for (int r = 0; r < 8; ++r) {
        const int m = m0 + wm * 32 + mi * 16 + r + 8 * half;
        const int n = n0 + wn * 32 + ni * 16 + ln;
        if (MODE == 0) {
          const int b = m >> 10, l = m & 1023, h = n >> 6, d = n & 63;
          ((__bf16*)Out)[((((size_t)b * NH) + h) * SEQ + l) * DK + d] = (__bf16)acc[mi][ni][r];
        } else {
          ((float*)Out)[(size_t)m * DM + n] = acc[mi][ni][r];
        }
      }
}

// ---------------- Fused Shaw attention (bh-grouped flash) ----------------
struct SmemAtt {
  __bf16 Qs[HG][IT][DK];                                   // 16 KB
  __bf16 Ks[HG][JT][DK];                                   // 16 KB
  __bf16 Vs[HG][JT][DK];                                   // 16 KB
  union { __bf16 RKs[IT][JT][DK]; float stage[IT][HG][DK]; };  // 32 KB
  __bf16 RVs[IT][JT][DK];                                  // 32 KB
  float  Sr[IT][HG][JT];                                   //  8 KB
  __bf16 P[HG][IT][JT];                                    //  4 KB
  float  fac[NH][IT];                                      //  1 KB
};  // total 128000 B

__global__ __launch_bounds__(512) void attn_kernel(
    const __bf16* __restrict__ Qg, const __bf16* __restrict__ Kg,
    const __bf16* __restrict__ Vg, const int* __restrict__ relIdx,
    const float* __restrict__ relK, const float* __restrict__ relV,
    __bf16* __restrict__ AttnOut) {
  __shared__ SmemAtt sm;
  const int t = threadIdx.x, lane = t & 31, w = t >> 5;
  const int half = lane >> 4, ln = lane & 15;
  const int i0 = blockIdx.x * IT;
  const int g = blockIdx.y;
  const int b = g >> 1, h0 = (g & 1) * HG;
  const float scale = 0.125f;  // 1/sqrt(64)

  // neutral factors for empty head rows (8..15) of the O_rel accumulators
  if (t < (NH - HG) * IT) sm.fac[HG + t / IT][t % IT] = 1.0f;

  {  // Q tile: [HG][IT][DK]  (async global->LDS when available)
    const int row = t >> 2, dseg = (t & 3) * 16;
    const int h = row >> 4, i = row & 15;
    const __bf16* src = Qg + ((((size_t)b * NH) + h0 + h) * SEQ + i0 + i) * DK + dseg;
    cp16(&sm.Qs[h][i][dseg], src);
    cp16(&sm.Qs[h][i][dseg + 8], src + 8);
  }

  v8f oc[4] = {}, orr[4] = {};   // P@V (per head w<8) and P@r_v (per i=w) accumulators
  float mrow[8], lrow[8];
  #pragma unroll
  for (int r = 0; r < 8; ++r) { mrow[r] = -INFINITY; lrow[r] = 0.0f; }

  for (int j0 = 0; j0 < SEQ; j0 += JT) {
    __syncthreads();
    {  // K,V tiles (async global->LDS)
      const int row = t >> 2, dseg = (t & 3) * 16;
      const int h = row >> 4, j = row & 15;
      const size_t off = ((((size_t)b * NH) + h0 + h) * SEQ + j0 + j) * DK + dseg;
      cp16(&sm.Ks[h][j][dseg],     Kg + off);
      cp16(&sm.Ks[h][j][dseg + 8], Kg + off + 8);
      cp16(&sm.Vs[h][j][dseg],     Vg + off);
      cp16(&sm.Vs[h][j][dseg + 8], Vg + off + 8);
      if (j0 + JT < SEQ) {  // prefetch next tile's K/V lines
        const size_t noff = off + (size_t)JT * DK;
        __builtin_prefetch(Kg + noff, 0, 1);
        __builtin_prefetch(Vg + noff, 0, 1);
      }
    }
    {  // gathered rel_k / rel_v tiles (shared across the 8 heads -> GEMM-able)
      const int pair = t >> 1, dh = (t & 1) * 32;
      const int i = pair >> 4, j = pair & 15;
      const int idx = relIdx[(size_t)(i0 + i) * SEQ + j0 + j];
      const float* sk = relK + (size_t)idx * DK + dh;
      const float* sv = relV + (size_t)idx * DK + dh;
      #pragma unroll
      for (int q = 0; q < 32; ++q) {
        sm.RKs[i][j][dh + q] = (__bf16)sk[q];
        sm.RVs[i][j][dh + q] = (__bf16)sv[q];
      }
      if (j0 + JT < SEQ)  // prefetch next tile's index row
        __builtin_prefetch(&relIdx[(size_t)(i0 + i) * SEQ + j0 + JT + j], 0, 1);
    }
    cp_wait_all();   // s_wait_asynccnt 0 (barrier does not cover ASYNCcnt)
    __syncthreads();

    // rel scores: wave w owns query row i=w; M-axis = heads (GEMM over bh!)
    v8f crel = {};
    #pragma unroll
    for (int kb = 0; kb < 2; ++kb) {
      v16bf a  = frag_a(&sm.Qs[0][w][kb * 32], IT * DK, HG, 32);
      v16bf bb = frag_b_nk(&sm.RKs[w][0][kb * 32], DK, 32);
      crel = WMMA_BF16(a, bb, crel);
    }
    // content scores: wave w<8 owns head h=w; M-axis = query rows
    v8f ccon = {};
    if (w < HG) {
      #pragma unroll
      for (int kb = 0; kb < 2; ++kb) {
        v16bf a  = frag_a(&sm.Qs[w][0][kb * 32], DK, 16, 32);
        v16bf bb = frag_b_nk(&sm.Ks[w][0][kb * 32], DK, 32);
        ccon = WMMA_BF16(a, bb, ccon);
      }
    }
    if (lane < 16) {  // rows h=r (half 0) are the valid 8
      #pragma unroll
      for (int r = 0; r < 8; ++r) sm.Sr[w][r][lane] = crel[r];
    }
    __syncthreads();

    // online softmax (head-waves), row i = r + 8*half, 16 j's across lanes of a half
    if (w < HG) {
      #pragma unroll
      for (int r = 0; r < 8; ++r) {
        const int i = r + 8 * half;
        float s = (ccon[r] + sm.Sr[i][w][ln]) * scale;
        float mx = s;
        #pragma unroll
        for (int off = 1; off < 16; off <<= 1) mx = fmaxf(mx, __shfl_xor(mx, off, 32));
        const float mnew = fmaxf(mrow[r], mx);
        const float p = __expf(s - mnew);
        float sum = p;
        #pragma unroll
        for (int off = 1; off < 16; off <<= 1) sum += __shfl_xor(sum, off, 32);
        const float f = __expf(mrow[r] - mnew);
        lrow[r] = lrow[r] * f + sum;
        mrow[r] = mnew;
        #pragma unroll
        for (int nt = 0; nt < 4; ++nt) oc[nt][r] *= f;
        sm.P[w][i][ln] = (__bf16)p;
        if (ln == 0) sm.fac[w][i] = f;
      }
    }
    __syncthreads();

    // rescale + accumulate P @ r_v  (wave w owns i=w; M-axis = heads)
    #pragma unroll
    for (int r = 0; r < 8; ++r) {
      const float f = sm.fac[r + 8 * half][w];
      #pragma unroll
      for (int nt = 0; nt < 4; ++nt) orr[nt][r] *= f;
    }
    {
      v16bf arv = frag_a(&sm.P[0][w][0], IT * JT, HG, JT);
      #pragma unroll
      for (int nt = 0; nt < 4; ++nt) {
        v16bf bb = frag_b_kn(&sm.RVs[w][0][nt * 16], DK, JT);
        orr[nt] = WMMA_BF16(arv, bb, orr[nt]);
      }
    }
    // accumulate P @ V (head-waves)
    if (w < HG) {
      v16bf apv = frag_a(&sm.P[w][0][0], JT, 16, JT);
      #pragma unroll
      for (int nt = 0; nt < 4; ++nt) {
        v16bf bb = frag_b_kn(&sm.Vs[w][0][nt * 16], DK, JT);
        oc[nt] = WMMA_BF16(apv, bb, oc[nt]);
      }
    }
  }

  __syncthreads();
  // i-waves publish O_rel (rows h<8 live in half 0) into stage (overlaps RKs)
  if (lane < 16) {
    #pragma unroll
    for (int nt = 0; nt < 4; ++nt)
      #pragma unroll
      for (int r = 0; r < 8; ++r) sm.stage[w][r][nt * 16 + lane] = orr[nt][r];
  }
  __syncthreads();
  // head-waves combine, normalize, store bf16 [B][SEQ][DM]
  if (w < HG) {
    #pragma unroll
    for (int nt = 0; nt < 4; ++nt)
      #pragma unroll
      for (int r = 0; r < 8; ++r) {
        const int i = r + 8 * half, d = nt * 16 + ln;
        const float val = (oc[nt][r] + sm.stage[i][w][d]) / lrow[r];
        AttnOut[((size_t)b * SEQ + i0 + i) * DM + (h0 + w) * DK + d] = (__bf16)val;
      }
  }
}

// ---------------- launcher ----------------
extern "C" void kernel_launch(void* const* d_in, const int* in_sizes, int n_in,
                              void* d_out, int out_size, void* d_ws, size_t ws_size,
                              hipStream_t stream) {
  const float* x      = (const float*)d_in[0];
  const int*   relIdx = (const int*)d_in[1];
  const float* Wq     = (const float*)d_in[2];
  const float* Wk     = (const float*)d_in[3];
  const float* Wv     = (const float*)d_in[4];
  const float* Wo     = (const float*)d_in[5];
  const float* relK   = (const float*)d_in[6];
  const float* relV   = (const float*)d_in[7];
  float* out = (float*)d_out;

  char* ws = (char*)d_ws;
  const size_t qkvBytes = (size_t)BATCH * NH * SEQ * DK * sizeof(__bf16);  // 16 MB each
  __bf16* qg   = (__bf16*)(ws + 0 * qkvBytes);
  __bf16* kg   = (__bf16*)(ws + 1 * qkvBytes);
  __bf16* vg   = (__bf16*)(ws + 2 * qkvBytes);
  __bf16* attn = (__bf16*)(ws + 3 * qkvBytes);  // [B][SEQ][DM] bf16, 16 MB

  dim3 gg(DM / 64, (BATCH * SEQ) / 128);
  gemm_bf16<float, 0><<<gg, 256, 0, stream>>>(x, Wq, qg);
  gemm_bf16<float, 0><<<gg, 256, 0, stream>>>(x, Wk, kg);
  gemm_bf16<float, 0><<<gg, 256, 0, stream>>>(x, Wv, vg);

  dim3 ga(SEQ / IT, (BATCH * NH) / HG);
  attn_kernel<<<ga, 512, 0, stream>>>(qg, kg, vg, relIdx, relK, relV, attn);

  gemm_bf16<__bf16, 1><<<gg, 256, 0, stream>>>(attn, Wo, out);
}
